// TransitionGNN_60138132078980
// MI455X (gfx1250) — matrix operand order repaired and checked by reading, not compiled, not run on hardware
//
#include <hip/hip_runtime.h>

// ---------------------------------------------------------------------------
// TransitionGNN fused kernel for gfx1250 (MI455X), bf16 WMMA + fp32 accum.
// B=4096 batches, K=16 nodes, D=32, H=128, A=32. 240 edges/batch (i!=j).
// One workgroup (256 threads = 8 wave32) processes 8 batch elements,
// keeping all weights (bf16, transposed to [N][K]) + activations in LDS.
// Multi-accumulator WMMA chains avoid the D->C data-hazard NOPs.
// ---------------------------------------------------------------------------

#define NB    4096
#define BPB   8      // batches per block

typedef __bf16 bf16;
typedef __attribute__((ext_vector_type(16))) __bf16 v16bf;
typedef __attribute__((ext_vector_type(8)))  __bf16 v8bf;
typedef __attribute__((ext_vector_type(8)))  float  v8f;

struct SmemT {
  // weights, bf16, stored transposed: Wt[n*Kdim + k]
  bf16 We1[128 * 64];    // ew1: K=64,  N=128
  bf16 We2[128 * 128];   // ew2
  bf16 We3[128 * 128];   // ew3
  bf16 Wn1[128 * 192];   // nw1: K=192, N=128
  bf16 Wn2[128 * 128];   // nw2
  bf16 Wn3[32 * 128];    // nw3: K=128, N=32
  // activations (bf16)
  bf16 Xb[16 * 32];      // node block, bf16
  bf16 Atile[16 * 192];  // A staging (edge 16x64 / node 16x192), ld=192
  bf16 Htile[16 * 128];  // layer-1 output (post relu)
  bf16 H2[16 * 128];     // post-LN relu bf16
  // fp32 work buffers
  float Tf[16 * 128];    // pre-LN fp32
  float Agg[16 * 128];   // scatter-add accumulator per batch
  float Psum[16 * 16], Psum2[16 * 16];
  float Mu[16], Ri[16];
  // biases / LN params (fp32)
  float be1[128], be2[128], be3[128];
  float bn1[128], bn2[128], bn3[32];
  float egs[128], ebts[128], ngs[128], nbts[128];
};

__device__ __forceinline__ bf16 f2bf(float f) {
  unsigned u = __builtin_bit_cast(unsigned, f);
  u += 0x7FFFu + ((u >> 16) & 1u);               // round-to-nearest-even
  unsigned short h = (unsigned short)(u >> 16);
  return __builtin_bit_cast(bf16, h);
}

// A fragment (16x32 bf16, M=row per lane): lanes 0-15 row M=l hold K[k0..k0+15];
// lanes 16-31 hold K[k0+16..k0+31].
__device__ __forceinline__ v16bf load_a(const bf16* base, int ldk, int k0, int lane) {
  int row = lane & 15, half = lane >> 4;
  const bf16* p = base + row * ldk + k0 + 16 * half;
  v8bf lo = *(const v8bf*)p;
  v8bf hi = *(const v8bf*)(p + 8);
  return __builtin_shufflevector(lo, hi, 0, 1, 2, 3, 4, 5, 6, 7,
                                 8, 9, 10, 11, 12, 13, 14, 15);
}

// B fragment (32x16 bf16, column N per lane): lane n<16 -> col n0+n, K[k0..k0+15];
// lane n+16 -> col n0+n, K[k0+16..k0+31]. Weights stored transposed [N][K].
__device__ __forceinline__ v16bf load_b(const bf16* wt, int ldk, int n0, int k0, int lane) {
  int col = n0 + (lane & 15), half = lane >> 4;
  return *(const v16bf*)(wt + col * ldk + k0 + 16 * half);
}

__device__ __forceinline__ v8f wmma_bf16(v16bf a, v16bf b, v8f c) {
  return __builtin_amdgcn_wmma_f32_16x16x32_bf16(false, a, false, b,
                                                 (short)0, c, false, false);
}

// KS-step GEMM tile with dual accumulators: consecutive WMMAs are independent,
// avoiding the D->C chained-accumulation hazard NOPs (cdna5 §7.12.1).
template <int KS>
__device__ __forceinline__ v8f gemm_tile(const bf16* Abase, int ldk,
                                         const v16bf* Bf, int lane) {
  const v8f vz = {0.f, 0.f, 0.f, 0.f, 0.f, 0.f, 0.f, 0.f};
  v8f c0 = vz, c1 = vz;
#pragma unroll
  for (int kk = 0; kk < KS; ++kk) {
    v16bf a = load_a(Abase, ldk, kk * 32, lane);
    if (kk & 1) c1 = wmma_bf16(a, Bf[kk], c1);
    else        c0 = wmma_bf16(a, Bf[kk], c0);
  }
  if (KS > 1) {
#pragma unroll
    for (int q = 0; q < 8; ++q) c0[q] += c1[q];
  }
  return c0;
}

// LayerNorm(+relu, +bf16 cast) over Tf[16][128] -> H2, fp32 stats.
__device__ __forceinline__ void ln_relu_128(SmemT* sm, const float* g,
                                            const float* bt, int tid) {
  {
    int rowp = tid >> 4, seg = tid & 15;
    const float* tr = sm->Tf + rowp * 128 + seg * 8;
    float s = 0.f, s2 = 0.f;
#pragma unroll
    for (int q = 0; q < 8; ++q) { float v = tr[q]; s += v; s2 += v * v; }
    sm->Psum[rowp * 16 + seg]  = s;
    sm->Psum2[rowp * 16 + seg] = s2;
  }
  __syncthreads();
  if (tid < 16) {
    float s = 0.f, s2 = 0.f;
#pragma unroll
    for (int q = 0; q < 16; ++q) {
      s  += sm->Psum[tid * 16 + q];
      s2 += sm->Psum2[tid * 16 + q];
    }
    float mu  = s * 0.0078125f;
    float var = s2 * 0.0078125f - mu * mu;
    sm->Mu[tid] = mu;
    sm->Ri[tid] = rsqrtf(var + 1e-5f);
  }
  __syncthreads();
  for (int idx = tid; idx < 2048; idx += 256) {
    int rr = idx >> 7, cc = idx & 127;
    float v = (sm->Tf[idx] - sm->Mu[rr]) * sm->Ri[rr] * g[cc] + bt[cc];
    sm->H2[idx] = f2bf(fmaxf(v, 0.f));
  }
  __syncthreads();
}

extern __shared__ __align__(128) char smem_raw[];

__global__ void __launch_bounds__(256, 1)
gnn_kernel(const float* __restrict__ states, const int* __restrict__ action,
           const float* __restrict__ ew1, const float* __restrict__ eb1,
           const float* __restrict__ ew2, const float* __restrict__ eb2,
           const float* __restrict__ eg,  const float* __restrict__ ebt,
           const float* __restrict__ ew3, const float* __restrict__ eb3,
           const float* __restrict__ nw1, const float* __restrict__ nb1,
           const float* __restrict__ nw2, const float* __restrict__ nb2,
           const float* __restrict__ ng,  const float* __restrict__ nbt,
           const float* __restrict__ nw3, const float* __restrict__ nb3,
           float* __restrict__ out) {
  SmemT* sm = reinterpret_cast<SmemT*>(smem_raw);
  const int tid  = threadIdx.x;
  const int wave = tid >> 5;
  const int lane = tid & 31;
  const int half = lane >> 4;
  const int n0   = wave * 16;            // output-column tile per wave
  const int colX = n0 + (lane & 15);     // this lane's output column (N<=128)

  // ---- stage weights into LDS, bf16, transposed [N][K] --------------------
  {
    auto ldW = [&](const float* W, bf16* Wt, int Kd, int N) {
      for (int idx = tid; idx < Kd * N; idx += 256) {
        int k = idx / N, n = idx - k * N;
        Wt[n * Kd + k] = f2bf(W[idx]);
      }
    };
    ldW(ew1, sm->We1, 64, 128);
    ldW(ew2, sm->We2, 128, 128);
    ldW(ew3, sm->We3, 128, 128);
    ldW(nw1, sm->Wn1, 192, 128);
    ldW(nw2, sm->Wn2, 128, 128);
    ldW(nw3, sm->Wn3, 128, 32);
    for (int i = tid; i < 128; i += 256) {
      sm->be1[i] = eb1[i]; sm->be2[i] = eb2[i]; sm->be3[i] = eb3[i];
      sm->bn1[i] = nb1[i]; sm->bn2[i] = nb2[i];
      sm->egs[i] = eg[i];  sm->ebts[i] = ebt[i];
      sm->ngs[i] = ng[i];  sm->nbts[i] = nbt[i];
    }
    if (tid < 32) sm->bn3[tid] = nb3[tid];
  }
  __syncthreads();

  for (int bi = 0; bi < BPB; ++bi) {
    const int b = blockIdx.x * BPB + bi;

    // prefetch next batch's node block while we compute this one
    if (bi + 1 < BPB && lane < 16)
      __builtin_prefetch(states + (b + 1) * 512 + lane * 32, 0, 1);

    // ---- per-batch init: node block (bf16) + zero Agg ---------------------
    for (int idx = tid; idx < 2048; idx += 256) sm->Agg[idx] = 0.f;
    for (int idx = tid; idx < 512; idx += 256)
      sm->Xb[idx] = f2bf(states[b * 512 + idx]);
    const int act = action[b];
    __syncthreads();

    // edge-layer B fragments live across the M-tile loop
    v16bf Be1[2], Be2[4], Be3[4];
#pragma unroll
    for (int k2 = 0; k2 < 2; ++k2) Be1[k2] = load_b(sm->We1, 64, n0, k2 * 32, lane);
#pragma unroll
    for (int k2 = 0; k2 < 4; ++k2) Be2[k2] = load_b(sm->We2, 128, n0, k2 * 32, lane);
#pragma unroll
    for (int k2 = 0; k2 < 4; ++k2) Be3[k2] = load_b(sm->We3, 128, n0, k2 * 32, lane);

    // ================= edge MLP over 15 tiles of 16 edges ==================
    for (int mt = 0; mt < 15; ++mt) {
      // build e_in tile [16 x 64] = [x_src | x_dst]; ld = 192
      for (int idx = tid; idx < 1024; idx += 256) {
        int rr = idx >> 6, cc = idx & 63;
        int e = mt * 16 + rr;
        int i = e / 15;
        int jj = e - i * 15;
        int j = jj + (jj >= i ? 1 : 0);
        sm->Atile[rr * 192 + cc] =
            (cc < 32) ? sm->Xb[i * 32 + cc] : sm->Xb[j * 32 + cc - 32];
      }
      __syncthreads();

      // e1: [16x64]@[64x128] + bias, relu -> Htile (bf16)
      {
        v8f c = gemm_tile<2>(sm->Atile, 192, Be1, lane);
        float bv = sm->be1[colX];
#pragma unroll
        for (int r = 0; r < 8; ++r) {
          int row = r + 8 * half;
          sm->Htile[row * 128 + colX] = f2bf(fmaxf(c[r] + bv, 0.f));
        }
      }
      __syncthreads();

      // e2: [16x128]@[128x128] + bias -> Tf (fp32), then LN+relu -> H2
      {
        v8f c = gemm_tile<4>(sm->Htile, 128, Be2, lane);
        float bv = sm->be2[colX];
#pragma unroll
        for (int r = 0; r < 8; ++r) {
          int row = r + 8 * half;
          sm->Tf[row * 128 + colX] = c[r] + bv;
        }
      }
      __syncthreads();
      ln_relu_128(sm, sm->egs, sm->ebts, tid);

      // e3: [16x128]@[128x128] + bias, scatter-add into Agg[src]
      {
        v8f c = gemm_tile<4>(sm->H2, 128, Be3, lane);
        float bv = sm->be3[colX];
#pragma unroll
        for (int r = 0; r < 8; ++r) {
          int row = r + 8 * half;
          int e = mt * 16 + row;
          int i = e / 15;                 // source node of this edge
          atomicAdd(&sm->Agg[i * 128 + colX], c[r] + bv);
        }
      }
      __syncthreads();
    }

    // ======================== node MLP (16 rows) ===========================
    // n_in = [X(32) | onehot(32) | Agg(128)] as bf16, ld = 192
    for (int idx = tid; idx < 16 * 192; idx += 256) {
      int rr = idx / 192, cc = idx - rr * 192;
      bf16 v;
      if (cc < 32) {
        v = sm->Xb[rr * 32 + cc];
      } else if (cc < 64) {
        v = f2bf(((act >> 5) == rr && (act & 31) == (cc - 32)) ? 1.f : 0.f);
      } else {
        v = f2bf(sm->Agg[rr * 128 + (cc - 64)]);
      }
      sm->Atile[idx] = v;
    }
    __syncthreads();

    // n1: [16x192]@[192x128] + bias, relu -> Htile
    {
      v16bf Bn1[6];
#pragma unroll
      for (int k2 = 0; k2 < 6; ++k2) Bn1[k2] = load_b(sm->Wn1, 192, n0, k2 * 32, lane);
      v8f c = gemm_tile<6>(sm->Atile, 192, Bn1, lane);
      float bv = sm->bn1[colX];
#pragma unroll
      for (int r = 0; r < 8; ++r) {
        int row = r + 8 * half;
        sm->Htile[row * 128 + colX] = f2bf(fmaxf(c[r] + bv, 0.f));
      }
    }
    __syncthreads();

    // n2: [16x128]@[128x128] + bias -> Tf, LN+relu -> H2
    {
      v16bf Bn2[4];
#pragma unroll
      for (int k2 = 0; k2 < 4; ++k2) Bn2[k2] = load_b(sm->Wn2, 128, n0, k2 * 32, lane);
      v8f c = gemm_tile<4>(sm->Htile, 128, Bn2, lane);
      float bv = sm->bn2[colX];
#pragma unroll
      for (int r = 0; r < 8; ++r) {
        int row = r + 8 * half;
        sm->Tf[row * 128 + colX] = c[r] + bv;
      }
    }
    __syncthreads();
    ln_relu_128(sm, sm->ngs, sm->nbts, tid);

    // n3: [16x128]@[128x32] + bias -> out (fp32). Waves 0,1 cover N=32.
    if (wave < 2) {
      v16bf Bn3[4];
#pragma unroll
      for (int k2 = 0; k2 < 4; ++k2) Bn3[k2] = load_b(sm->Wn3, 128, n0, k2 * 32, lane);
      v8f c = gemm_tile<4>(sm->H2, 128, Bn3, lane);
      int col = n0 + (lane & 15);
      float bv = sm->bn3[col];
#pragma unroll
      for (int r = 0; r < 8; ++r) {
        int row = r + 8 * half;
        out[(b * 16 + row) * 32 + col] = c[r] + bv;
      }
    }
    __syncthreads();
  }
}

extern "C" void kernel_launch(void* const* d_in, const int* in_sizes, int n_in,
                              void* d_out, int out_size, void* d_ws, size_t ws_size,
                              hipStream_t stream) {
  (void)in_sizes; (void)n_in; (void)out_size; (void)d_ws; (void)ws_size;
  const float* states = (const float*)d_in[0];
  const int*   action = (const int*)d_in[1];
  // d_in[2] = edge_index (unused: graph pattern is fixed complete-minus-diag)
  const float* ew1 = (const float*)d_in[3];
  const float* eb1 = (const float*)d_in[4];
  const float* ew2 = (const float*)d_in[5];
  const float* eb2 = (const float*)d_in[6];
  const float* eg  = (const float*)d_in[7];
  const float* ebt = (const float*)d_in[8];
  const float* ew3 = (const float*)d_in[9];
  const float* eb3 = (const float*)d_in[10];
  const float* nw1 = (const float*)d_in[11];
  const float* nb1 = (const float*)d_in[12];
  const float* nw2 = (const float*)d_in[13];
  const float* nb2 = (const float*)d_in[14];
  const float* ng  = (const float*)d_in[15];
  const float* nbt = (const float*)d_in[16];
  const float* nw3 = (const float*)d_in[17];
  const float* nb3 = (const float*)d_in[18];
  float* out = (float*)d_out;

  const int smem = (int)sizeof(SmemT);  // ~206 KB; gfx1250 WGP has 320 KB LDS
  hipFuncSetAttribute(reinterpret_cast<const void*>(gnn_kernel),
                      hipFuncAttributeMaxDynamicSharedMemorySize, smem);
  gnn_kernel<<<NB / BPB, 256, smem, stream>>>(
      states, action, ew1, eb1, ew2, eb2, eg, ebt, ew3, eb3,
      nw1, nb1, nw2, nb2, ng, nbt, nw3, nb3, out);
}
